// SequenceBlock_3315714752611
// MI455X (gfx1250) — matrix operand order repaired and checked by reading, not compiled, not run on hardware
//
#include <hip/hip_runtime.h>
#include <math.h>

#define LQ 4096
#define HQ 256
#define NQ 64
#define TQ 128          // chunk length
#define NCQ 32          // LQ / TQ

typedef __attribute__((ext_vector_type(16))) _Float16 v16h;
typedef __attribute__((ext_vector_type(8)))  _Float16 v8h;
typedef __attribute__((ext_vector_type(8)))  float    v8f;
typedef __attribute__((ext_vector_type(4)))  float    v4f;

static __device__ __forceinline__ v16h cat16(v8h lo, v8h hi) {
  v16h r;
#pragma unroll
  for (int i = 0; i < 8; ++i) { r[i] = lo[i]; r[i + 8] = hi[i]; }
  return r;
}

// Fast reciprocal: single v_rcp_f32 (not the IEEE div expansion)
static __device__ __forceinline__ float rcp_fast(float v) {
  return __builtin_amdgcn_rcpf(v);
}

// Branch-free gelu (tanh approx): 0.5*v*(1+tanh(z)) == v * sigmoid(2z)
static __device__ __forceinline__ float gelu_f(float v) {
  const float z2 = 1.5957691216057308f * (v + 0.044715f * v * v * v); // 2*sqrt(2/pi)*(...)
  return v * rcp_fast(1.0f + __expf(-z2));
}

static __device__ __forceinline__ float sigmoid_f(float v) {
  return rcp_fast(1.0f + __expf(-v));
}

// ---------------------------------------------------------------------------
// Kernel 1: per-token LayerNorm over H (one block per token, H threads)
// ---------------------------------------------------------------------------
__global__ __launch_bounds__(256) void ln_kernel(const float* __restrict__ x,
                                                 const float* __restrict__ w,
                                                 const float* __restrict__ b,
                                                 float* __restrict__ xn) {
  __shared__ float redA[8];
  __shared__ float redB[8];
  const int l = blockIdx.x;
  const int tid = threadIdx.x;
  const float v = x[(size_t)l * HQ + tid];
  float s = v;
#pragma unroll
  for (int o = 16; o > 0; o >>= 1) s += __shfl_xor(s, o, 32);
  if ((tid & 31) == 0) redA[tid >> 5] = s;
  __syncthreads();
  float mu = 0.f;
#pragma unroll
  for (int i = 0; i < 8; ++i) mu += redA[i];
  mu *= (1.0f / HQ);
  const float d = v - mu;
  float q = d * d;
#pragma unroll
  for (int o = 16; o > 0; o >>= 1) q += __shfl_xor(q, o, 32);
  if ((tid & 31) == 0) redB[tid >> 5] = q;
  __syncthreads();
  float var = 0.f;
#pragma unroll
  for (int i = 0; i < 8; ++i) var += redB[i];
  var *= (1.0f / HQ);
  xn[(size_t)l * HQ + tid] =
      d * __builtin_amdgcn_rsqf(var + 1e-5f) * w[tid] + b[tid];
}

// ---------------------------------------------------------------------------
// Kernel 2 (scan phase A): per-(chunk, head) local scan with zero init state.
// One wave per (chunk, head); each lane owns states n=lane and n=lane+32.
// ---------------------------------------------------------------------------
__global__ __launch_bounds__(128) void scanA(
    const float* __restrict__ xn,
    const float* __restrict__ Lre, const float* __restrict__ Lim,
    const float* __restrict__ Bre, const float* __restrict__ Bim,
    const float* __restrict__ Cre, const float* __restrict__ Cim,
    const float* __restrict__ Dp,  const float* __restrict__ log_step,
    float* __restrict__ ys, float* __restrict__ cs) {
  const int lane = threadIdx.x & 31;
  const int wid  = threadIdx.x >> 5;
  const int c = blockIdx.x;               // chunk
  const int h = blockIdx.y * 4 + wid;     // head
  const float dt = __expf(log_step[h]);

  float ar[2], ai[2], br[2], bi[2], cr[2], ci[2];
#pragma unroll
  for (int j = 0; j < 2; ++j) {
    const int n = lane + 32 * j;
    const float lr = Lre[h * NQ + n], li = Lim[h * NQ + n];
    const float e = __expf(lr * dt);
    const float abre = e * __cosf(li * dt);
    const float abim = e * __sinf(li * dt);
    const float nr = abre - 1.0f, ni = abim;
    const float inv = rcp_fast(lr * lr + li * li);
    const float qr = (nr * lr + ni * li) * inv;
    const float qi = (ni * lr - nr * li) * inv;
    const float Br = Bre[h * NQ + n], Bi = Bim[h * NQ + n];
    ar[j] = abre; ai[j] = abim;
    br[j] = qr * Br - qi * Bi;
    bi[j] = qr * Bi + qi * Br;
    cr[j] = Cre[h * NQ + n]; ci[j] = Cim[h * NQ + n];
  }
  const float Dh = Dp[h];
  float s0r = 0.f, s0i = 0.f, s1r = 0.f, s1i = 0.f;
  const int l0 = c * TQ;

  for (int qblk = 0; qblk < TQ / 32; ++qblk) {
    const float uvec = xn[(size_t)(l0 + qblk * 32 + lane) * HQ + h];
    float yv = 0.f;
#pragma unroll
    for (int tt = 0; tt < 32; ++tt) {
      const float u = __shfl(uvec, tt, 32);
      const float t0r = ar[0] * s0r - ai[0] * s0i + br[0] * u;
      const float t0i = ar[0] * s0i + ai[0] * s0r + bi[0] * u;
      const float t1r = ar[1] * s1r - ai[1] * s1i + br[1] * u;
      const float t1i = ar[1] * s1i + ai[1] * s1r + bi[1] * u;
      s0r = t0r; s0i = t0i; s1r = t1r; s1i = t1i;
      float p = cr[0] * s0r - ci[0] * s0i + cr[1] * s1r - ci[1] * s1i;
#pragma unroll
      for (int o = 16; o > 0; o >>= 1) p += __shfl_xor(p, o, 32);
      const float y = 2.0f * p + Dh * u;
      yv = (tt == lane) ? y : yv;              // lane keeps its own timestep
    }
    ys[(size_t)(l0 + qblk * 32 + lane) * HQ + h] = yv;
  }
  float* cp = cs + ((size_t)c * HQ + h) * NQ * 2;
  cp[lane * 2] = s0r;        cp[lane * 2 + 1] = s0i;
  cp[(lane + 32) * 2] = s1r; cp[(lane + 32) * 2 + 1] = s1i;
}

// ---------------------------------------------------------------------------
// Kernel 3 (scan phase B): per-head sequential prefix over 32 chunk states.
// In-place: replaces cs[c] with s_in[c]. Writes final hidden state.
// ---------------------------------------------------------------------------
__global__ __launch_bounds__(128) void scanB(
    const float* __restrict__ Lre, const float* __restrict__ Lim,
    const float* __restrict__ log_step,
    float* __restrict__ cs, float* __restrict__ hidden) {
  const int lane = threadIdx.x & 31;
  const int wid  = threadIdx.x >> 5;
  const int h = blockIdx.x * 4 + wid;
  const float dt = __expf(log_step[h]);
  float aTr[2], aTi[2];
#pragma unroll
  for (int j = 0; j < 2; ++j) {
    const int n = lane + 32 * j;
    const float lr = Lre[h * NQ + n], li = Lim[h * NQ + n];
    const float e = __expf(lr * dt * (float)TQ);     // Ab^T = exp(Lam*dt*T)
    aTr[j] = e * __cosf(li * dt * (float)TQ);
    aTi[j] = e * __sinf(li * dt * (float)TQ);
  }
  float e0r = 0.f, e0i = 0.f, e1r = 0.f, e1i = 0.f;
  for (int c = 0; c < NCQ; ++c) {
    float* cp = cs + ((size_t)c * HQ + h) * NQ * 2;
    const float S0r = cp[lane * 2],        S0i = cp[lane * 2 + 1];
    const float S1r = cp[(lane + 32) * 2], S1i = cp[(lane + 32) * 2 + 1];
    cp[lane * 2] = e0r;        cp[lane * 2 + 1] = e0i;
    cp[(lane + 32) * 2] = e1r; cp[(lane + 32) * 2 + 1] = e1i;
    float t;
    t = aTr[0] * e0r - aTi[0] * e0i + S0r;
    e0i = aTr[0] * e0i + aTi[0] * e0r + S0i; e0r = t;
    t = aTr[1] * e1r - aTi[1] * e1i + S1r;
    e1i = aTr[1] * e1i + aTi[1] * e1r + S1i; e1r = t;
  }
  float* hp = hidden + (size_t)h * NQ * 2;
  hp[lane * 2] = e0r;        hp[lane * 2 + 1] = e0i;
  hp[(lane + 32) * 2] = e1r; hp[(lane + 32) * 2 + 1] = e1i;
}

// ---------------------------------------------------------------------------
// Kernel 4 (scan phase C): ys[l] += 2*Re( C . Ab^{t+1} s_in[c] ).
// ---------------------------------------------------------------------------
__global__ __launch_bounds__(128) void scanC(
    const float* __restrict__ cs,
    const float* __restrict__ Lre, const float* __restrict__ Lim,
    const float* __restrict__ Cre, const float* __restrict__ Cim,
    const float* __restrict__ log_step, float* __restrict__ ys) {
  const int lane = threadIdx.x & 31;
  const int wid  = threadIdx.x >> 5;
  const int c = blockIdx.x;
  const int h = blockIdx.y * 4 + wid;
  const float dt = __expf(log_step[h]);
  float ar[2], ai[2], cr[2], ci[2], tr[2], ti[2];
#pragma unroll
  for (int j = 0; j < 2; ++j) {
    const int n = lane + 32 * j;
    const float lr = Lre[h * NQ + n], li = Lim[h * NQ + n];
    const float e = __expf(lr * dt);
    ar[j] = e * __cosf(li * dt);
    ai[j] = e * __sinf(li * dt);
    cr[j] = Cre[h * NQ + n]; ci[j] = Cim[h * NQ + n];
  }
  const float* cp = cs + ((size_t)c * HQ + h) * NQ * 2;
  tr[0] = cp[lane * 2];        ti[0] = cp[lane * 2 + 1];
  tr[1] = cp[(lane + 32) * 2]; ti[1] = cp[(lane + 32) * 2 + 1];
  const int l0 = c * TQ;
  for (int qblk = 0; qblk < TQ / 32; ++qblk) {
    float cv = 0.f;
#pragma unroll
    for (int tt = 0; tt < 32; ++tt) {
      float t;
      t = ar[0] * tr[0] - ai[0] * ti[0];
      ti[0] = ar[0] * ti[0] + ai[0] * tr[0]; tr[0] = t;
      t = ar[1] * tr[1] - ai[1] * ti[1];
      ti[1] = ar[1] * ti[1] + ai[1] * tr[1]; tr[1] = t;
      float p = cr[0] * tr[0] - ci[0] * ti[0] + cr[1] * tr[1] - ci[1] * ti[1];
#pragma unroll
      for (int o = 16; o > 0; o >>= 1) p += __shfl_xor(p, o, 32);
      cv = (tt == lane) ? 2.0f * p : cv;
    }
    const size_t idx = (size_t)(l0 + qblk * 32 + lane) * HQ + h;
    ys[idx] += cv;
  }
}

// ---------------------------------------------------------------------------
// Kernel 5: fused gelu -> dual WMMA GEMM (g@W1^T, g@W2^T) -> bias ->
//           sigmoid gate -> skip add. 64x64 output tile per block, 8 waves.
// ---------------------------------------------------------------------------
__global__ __launch_bounds__(256) void glu_kernel(
    const float* __restrict__ ys, const float* __restrict__ x,
    const float* __restrict__ W1, const float* __restrict__ bb1,
    const float* __restrict__ W2, const float* __restrict__ bb2,
    float* __restrict__ out) {
  __shared__ _Float16 gS[64 * 40];     // 64 rows (l), 32 k + pad
  __shared__ _Float16 w1S[64 * 40];    // 64 rows (j), 32 k + pad (transposed B)
  __shared__ _Float16 w2S[64 * 40];

  const int tid  = threadIdx.x;
  const int lane = tid & 31;
  const int wid  = tid >> 5;
  const int lbase = blockIdx.x * 64;
  const int jbase = blockIdx.y * 64;
  const int rw = (wid & 3) * 16;       // wave's 16-row slice
  const int cw = (wid >> 2) * 32;      // wave's 32-col slice
  const int m    = lane & 15;
  const int half = lane >> 4;
  const int sr = tid >> 2;             // staging row   [0,64)
  const int sc = (tid & 3) * 8;        // staging col*8 {0,8,16,24}

  v8f acc1a = {}; v8f acc1b = {}; v8f acc2a = {}; v8f acc2b = {};

  for (int kt = 0; kt < 8; ++kt) {
    const int kb = kt * 32;
    {
      // Vectorized staging: 2x float4 per array, pack to v8h, one b128 store.
      const v4f* sg = (const v4f*)(ys + (size_t)(lbase + sr) * HQ + kb + sc);
      const v4f* s1 = (const v4f*)(W1 + (size_t)(jbase + sr) * HQ + kb + sc);
      const v4f* s2 = (const v4f*)(W2 + (size_t)(jbase + sr) * HQ + kb + sc);
      const v4f g0 = sg[0], g1 = sg[1];
      const v4f a0 = s1[0], a1 = s1[1];
      const v4f c0 = s2[0], c1 = s2[1];
      v8h hg, h1, h2;
#pragma unroll
      for (int i = 0; i < 4; ++i) {
        hg[i]     = (_Float16)gelu_f(g0[i]);
        hg[i + 4] = (_Float16)gelu_f(g1[i]);
        h1[i]     = (_Float16)a0[i];
        h1[i + 4] = (_Float16)a1[i];
        h2[i]     = (_Float16)c0[i];
        h2[i + 4] = (_Float16)c1[i];
      }
      *(v8h*)(&gS[sr * 40 + sc])  = hg;
      *(v8h*)(&w1S[sr * 40 + sc]) = h1;
      *(v8h*)(&w2S[sr * 40 + sc]) = h2;
    }
    __syncthreads();
    // A fragment: row M = rw+m; a[i] = A[M][16*half + i]
    const _Float16* ap = &gS[(rw + m) * 40];
    const v16h a = cat16(*(const v8h*)(ap + 16 * half),
                         *(const v8h*)(ap + 16 * half + 8));
    // B fragments: col N = cw(+16)+m; b[i] = B[16*half + i][N] = WT[N][16*half+i]
    const _Float16* p1a = &w1S[(cw + m) * 40 + 16 * half];
    const _Float16* p1b = &w1S[(cw + 16 + m) * 40 + 16 * half];
    const _Float16* p2a = &w2S[(cw + m) * 40 + 16 * half];
    const _Float16* p2b = &w2S[(cw + 16 + m) * 40 + 16 * half];
    const v16h B1a = cat16(*(const v8h*)(p1a), *(const v8h*)(p1a + 8));
    const v16h B1b = cat16(*(const v8h*)(p1b), *(const v8h*)(p1b + 8));
    const v16h B2a = cat16(*(const v8h*)(p2a), *(const v8h*)(p2a + 8));
    const v16h B2b = cat16(*(const v8h*)(p2b), *(const v8h*)(p2b + 8));

    acc1a = __builtin_amdgcn_wmma_f32_16x16x32_f16(false, a, false, B1a,
                                                   (short)0, acc1a, false, false);
    acc1b = __builtin_amdgcn_wmma_f32_16x16x32_f16(false, a, false, B1b,
                                                   (short)0, acc1b, false, false);
    acc2a = __builtin_amdgcn_wmma_f32_16x16x32_f16(false, a, false, B2a,
                                                   (short)0, acc2a, false, false);
    acc2b = __builtin_amdgcn_wmma_f32_16x16x32_f16(false, a, false, B2b,
                                                   (short)0, acc2b, false, false);
    __syncthreads();
  }
  // Epilogue: D-matrix element v -> row rw+v+8*half, col cw+m (+16)
#pragma unroll
  for (int v = 0; v < 8; ++v) {
    const int row = lbase + rw + v + 8 * half;
    const int c0 = jbase + cw + m;
    const int c1 = c0 + 16;
    const float a1 = acc1a[v] + bb1[c0];
    const float g1 = acc2a[v] + bb2[c0];
    out[(size_t)row * HQ + c0] = x[(size_t)row * HQ + c0] + a1 * sigmoid_f(g1);
    const float a2 = acc1b[v] + bb1[c1];
    const float g2 = acc2b[v] + bb2[c1];
    out[(size_t)row * HQ + c1] = x[(size_t)row * HQ + c1] + a2 * sigmoid_f(g2);
  }
}

// ---------------------------------------------------------------------------
extern "C" void kernel_launch(void* const* d_in, const int* in_sizes, int n_in,
                              void* d_out, int out_size, void* d_ws, size_t ws_size,
                              hipStream_t stream) {
  const float* x        = (const float*)d_in[0];
  const float* Lre      = (const float*)d_in[1];
  const float* Lim      = (const float*)d_in[2];
  const float* Bre      = (const float*)d_in[3];
  const float* Bim      = (const float*)d_in[4];
  const float* Cre      = (const float*)d_in[5];
  const float* Cim      = (const float*)d_in[6];
  const float* Dp       = (const float*)d_in[7];
  const float* log_step = (const float*)d_in[8];
  const float* ln_w     = (const float*)d_in[9];
  const float* ln_b     = (const float*)d_in[10];
  const float* W1       = (const float*)d_in[11];
  const float* b1       = (const float*)d_in[12];
  const float* W2       = (const float*)d_in[13];
  const float* b2       = (const float*)d_in[14];

  float* ws = (float*)d_ws;
  float* xn = ws;                                  // L*H
  float* ys = ws + (size_t)LQ * HQ;                // L*H
  float* cs = ws + 2 * (size_t)LQ * HQ;            // NC*H*N*2

  float* hidden = (float*)d_out;                   // H*N complex64 -> 2*H*N f32
  float* out2   = (float*)d_out + 2 * HQ * NQ;     // L*H f32

  ln_kernel<<<LQ, 256, 0, stream>>>(x, ln_w, ln_b, xn);
  scanA<<<dim3(NCQ, HQ / 4), 128, 0, stream>>>(xn, Lre, Lim, Bre, Bim, Cre, Cim,
                                               Dp, log_step, ys, cs);
  scanB<<<HQ / 4, 128, 0, stream>>>(Lre, Lim, log_step, cs, hidden);
  scanC<<<dim3(NCQ, HQ / 4), 128, 0, stream>>>(cs, Lre, Lim, Cre, Cim, log_step, ys);
  glu_kernel<<<dim3(LQ / 64, HQ / 64), 256, 0, stream>>>(ys, x, W1, b1, W2, b2, out2);
}
